// SelfAttention_20624432956365
// MI455X (gfx1250) — compile-verified, matrix operand order checked by reading, not run on hardware
//
#include <hip/hip_runtime.h>

// Problem constants (match reference)
constexpr int kB    = 4;
constexpr int kT    = 2048;
constexpr int kEMB  = 2048;
constexpr int kHD   = 128;
constexpr int kHALF = 64;

typedef __attribute__((ext_vector_type(16))) __bf16        v16bf;
typedef __attribute__((ext_vector_type(8)))  float         v8f;
typedef __attribute__((ext_vector_type(8)))  unsigned int  u32x8;
typedef __attribute__((ext_vector_type(4)))  unsigned int  u32x4;

// fp32 -> bf16 via native convert (lowers to v_cvt*_bf16_f32 on gfx1250)
__device__ inline unsigned short f2bf(float f) {
  return __builtin_bit_cast(unsigned short, (__bf16)f);
}
__device__ inline unsigned int pack2(float lo, float hi) {
  return (unsigned int)f2bf(lo) | ((unsigned int)f2bf(hi) << 16);
}

__device__ inline v8f wmma_bf16(v16bf a, v16bf b, v8f c) {
  // (neg_a, A, neg_b, B, c_mod, C, reuse_a, reuse_b)
  return __builtin_amdgcn_wmma_f32_16x16x32_bf16(false, a, false, b, (short)0, c,
                                                 false, false);
}

// 16x32 bf16 fragment load (A layout; B layout is identical with lane%16 = N col).
// rowp = start of this lane's row; per ISA the lane needs two contiguous 8-elem
// K-runs at offsets 8*half and 16+8*half -> two 16B loads.
__device__ inline v16bf ld_frag_bf16(const unsigned short* rowp, int half) {
  const u32x4 a = *(const u32x4*)(rowp + half * 8);
  const u32x4 b = *(const u32x4*)(rowp + 16 + half * 8);
  u32x8 r;
  r[0] = a[0]; r[1] = a[1]; r[2] = a[2]; r[3] = a[3];
  r[4] = b[0]; r[5] = b[1]; r[6] = b[2]; r[7] = b[3];
  return __builtin_bit_cast(v16bf, r);
}

// A-fragment built from fp32 source (x matrix), converting to bf16 on the fly.
__device__ inline v16bf ld_afrag_f32(const float* rowp, int half) {
  const int o1 = half * 8, o2 = 16 + half * 8;
  u32x8 r;
#pragma unroll
  for (int j = 0; j < 4; ++j) r[j]     = pack2(rowp[o1 + 2 * j], rowp[o1 + 2 * j + 1]);
#pragma unroll
  for (int j = 0; j < 4; ++j) r[4 + j] = pack2(rowp[o2 + 2 * j], rowp[o2 + 2 * j + 1]);
  return __builtin_bit_cast(v16bf, r);
}

// ---------------------------------------------------------------------------
// Kernel 1: Q/K/V projection (bf16 WMMA, fp32 accum) + RoPE, store bf16 to ws.
// Block = 256 threads = 8 waves; each wave owns one 16-row M tile; block shares
// a transposed bf16 W tile in LDS. grid.y selects Wq/Wk/Wv.
// ---------------------------------------------------------------------------
__global__ void __launch_bounds__(256) qkv_rope_kernel(
    const float* __restrict__ x, const float* __restrict__ Wq,
    const float* __restrict__ Wk, const float* __restrict__ Wv,
    const float* __restrict__ rot_cos, const float* __restrict__ rot_sin,
    unsigned short* __restrict__ qkv) {
  __shared__ __align__(16) unsigned short Wt[kHD * 32];  // [n][k] bf16

  const int tid  = threadIdx.x;
  const int lane = tid & 31;
  const int wave = tid >> 5;
  const int n16  = lane & 15;
  const int half = lane >> 4;

  const int proj = blockIdx.y;  // 0=Q 1=K 2=V
  const float* W = (proj == 0) ? Wq : (proj == 1) ? Wk : Wv;
  unsigned short* dst = qkv + (size_t)proj * (size_t)kB * kT * kHD;

  const int m0 = (blockIdx.x * 8 + wave) * 16;
  const float* xrow = x + (size_t)(m0 + n16) * kEMB;

  v8f acc[8];
#pragma unroll
  for (int nt = 0; nt < 8; ++nt)
#pragma unroll
    for (int i = 0; i < 8; ++i) acc[nt][i] = 0.0f;

  for (int kk = 0; kk < kEMB; kk += 32) {
    __syncthreads();
    {  // stage W[kk..kk+31][:] transposed into Wt[n][k] as bf16
      const int k  = tid >> 3;
      const int ng = (tid & 7) * 16;
      const float* wp = W + (size_t)(kk + k) * kHD + ng;
#pragma unroll
      for (int i = 0; i < 16; ++i) Wt[(ng + i) * 32 + k] = f2bf(wp[i]);
    }
    __syncthreads();

    const v16bf afrag = ld_afrag_f32(xrow + kk, half);
#pragma unroll
    for (int nt = 0; nt < 8; ++nt) {
      const v16bf bfrag = ld_frag_bf16(&Wt[(nt * 16 + n16) * 32], half);
      acc[nt] = wmma_bf16(afrag, bfrag, acc[nt]);
    }
  }

  // Epilogue: RoPE (Q,K only) then bf16 store. Lane holds (row = m0+8*half+r,
  // d = nt*16 + n16); RoPE pairs (2i,2i+1) live in adjacent lanes -> shfl_xor 1.
#pragma unroll
  for (int nt = 0; nt < 8; ++nt) {
#pragma unroll
    for (int r = 0; r < 8; ++r) {
      const int d   = nt * 16 + n16;
      const int row = m0 + half * 8 + r;
      float v = acc[nt][r];
      float res;
      if (proj < 2) {
        const float o  = __shfl_xor(v, 1);
        const int   t  = row & (kT - 1);
        const int   pr = d >> 1;
        const float c  = rot_cos[t * kHALF + pr];
        const float s  = rot_sin[t * kHALF + pr];
        res = (d & 1) ? (o * s + v * c) : (v * c - o * s);
      } else {
        res = v;
      }
      dst[(size_t)row * kHD + d] = f2bf(res);
    }
  }
}

// ---------------------------------------------------------------------------
// Kernel 2: causal flash attention. Block = 128 threads = 4 waves, each wave a
// 16-query tile; block shares K-tile (async-copied to LDS via the CDNA5
// GLOBAL_LOAD_ASYNC_TO_LDS path) and V-tile (transposed) in LDS.
// Online softmax in fp32; mask computed inline (mask input never read).
// ---------------------------------------------------------------------------
__global__ void __launch_bounds__(128) attention_kernel(
    const unsigned short* __restrict__ qkv, float* __restrict__ out) {
  __shared__ __align__(16) unsigned short Kt[32 * kHD];     // [key][d]
  __shared__ __align__(16) unsigned short Vt[kHD * 32];     // [d][key]
  __shared__ __align__(16) unsigned short Ps[4 * 16 * 32];  // per-wave P tile

  const int tid  = threadIdx.x;
  const int lane = tid & 31;
  const int wave = tid >> 5;
  const int n16  = lane & 15;
  const int half = lane >> 4;
  const int b    = blockIdx.y;
  const int q0   = (blockIdx.x * 4 + wave) * 16;

  const size_t plane = (size_t)kB * kT * kHD;
  const unsigned short* Qb = qkv + (size_t)b * kT * kHD;
  const unsigned short* Kb = qkv + plane + (size_t)b * kT * kHD;
  const unsigned short* Vb = qkv + 2 * plane + (size_t)b * kT * kHD;

  // Q fragments: 16 rows x 128 d, four K=32 chunks
  v16bf qfrag[4];
#pragma unroll
  for (int c = 0; c < 4; ++c)
    qfrag[c] = ld_frag_bf16(Qb + (size_t)(q0 + n16) * kHD + c * 32, half);

  v8f o[8];
#pragma unroll
  for (int nt = 0; nt < 8; ++nt)
#pragma unroll
    for (int i = 0; i < 8; ++i) o[nt][i] = 0.0f;
  float mrow[8], lrow[8];
#pragma unroll
  for (int r = 0; r < 8; ++r) { mrow[r] = -3.0e38f; lrow[r] = 0.0f; }

  const int   qlast = (blockIdx.x * 4 + 3) * 16 + 15;  // block-uniform bound
  const int   nkt   = qlast / 32 + 1;
  const float scale = 0.088388347648318447f;  // 1/sqrt(128)

  const unsigned ktLds = (unsigned)(uintptr_t)&Kt[0];  // LDS byte offset

  for (int kt = 0; kt < nkt; ++kt) {
    const int k0 = kt * 32;
    __syncthreads();
    {
      // K tile: 8 KB straight to LDS through the async-to-LDS path (no VGPR
      // round trip); tracked by ASYNCcnt.
      const unsigned long long kgbase =
          (unsigned long long)(uintptr_t)(Kb + (size_t)k0 * kHD);
#pragma unroll
      for (int i = 0; i < 4; ++i) {
        const int elem = tid + i * 128;                 // 512 x 16B chunks
        const unsigned lds = ktLds + (unsigned)elem * 16u;
        const unsigned long long ga = kgbase + (unsigned long long)elem * 16ull;
        asm volatile("global_load_async_to_lds_b128 %0, %1, off"
                     :: "v"(lds), "v"(ga) : "memory");
      }
      // V tile: transpose on the way into LDS (u32 global loads, coalesced).
      for (int i = tid; i < 32 * kHD / 2; i += 128) {
        const int key = i >> 6, dp = i & 63;            // dp = d/2
        const unsigned v =
            *(const unsigned*)(Vb + (size_t)(k0 + key) * kHD + dp * 2);
        Vt[(dp * 2) * 32 + key]     = (unsigned short)(v & 0xffffu);
        Vt[(dp * 2 + 1) * 32 + key] = (unsigned short)(v >> 16);
      }
      if (kt + 1 < nkt)
        __builtin_prefetch(Kb + (size_t)(k0 + 32) * kHD, 0, 0);
      asm volatile("s_wait_asynccnt 0x0" ::: "memory");
    }
    __syncthreads();

    if (k0 <= q0 + 15) {  // wave-uniform causal skip
      float Sv[2][8];
#pragma unroll
      for (int s2 = 0; s2 < 2; ++s2) {
        v8f S;
#pragma unroll
        for (int i = 0; i < 8; ++i) S[i] = 0.0f;
#pragma unroll
        for (int c = 0; c < 4; ++c) {
          const v16bf kf = ld_frag_bf16(&Kt[(s2 * 16 + n16) * kHD + c * 32], half);
          S = wmma_bf16(qfrag[c], kf, S);
        }
#pragma unroll
        for (int r = 0; r < 8; ++r) {
          float sv = S[r] * scale;
          const int kidx = k0 + s2 * 16 + n16;
          const int qidx = q0 + half * 8 + r;
          if (kidx > qidx) sv = -1.0e9f;
          Sv[s2][r] = sv;
        }
      }
      // online softmax: row stats via 16-lane butterflies (rows live per half)
      unsigned short* pw = Ps + wave * 512;
#pragma unroll
      for (int r = 0; r < 8; ++r) {
        float rmax = fmaxf(Sv[0][r], Sv[1][r]);
#pragma unroll
        for (int mm = 1; mm < 16; mm <<= 1) rmax = fmaxf(rmax, __shfl_xor(rmax, mm));
        const float mnew  = fmaxf(mrow[r], rmax);
        const float alpha = __expf(mrow[r] - mnew);
        const float p0 = __expf(Sv[0][r] - mnew);
        const float p1 = __expf(Sv[1][r] - mnew);
        float rsum = p0 + p1;
#pragma unroll
        for (int mm = 1; mm < 16; mm <<= 1) rsum += __shfl_xor(rsum, mm);
        lrow[r] = lrow[r] * alpha + rsum;
        mrow[r] = mnew;
#pragma unroll
        for (int nt = 0; nt < 8; ++nt) o[nt][r] *= alpha;
        const int m = half * 8 + r;  // reshape D-layout -> A-layout via LDS
        pw[m * 32 + n16]      = f2bf(p0);
        pw[m * 32 + 16 + n16] = f2bf(p1);
      }
      asm volatile("s_wait_dscnt 0x0" ::: "memory");  // same-wave LDS RAW
      const v16bf pfrag = ld_frag_bf16(pw + n16 * 32, half);
#pragma unroll
      for (int nt = 0; nt < 8; ++nt) {
        const v16bf vf = ld_frag_bf16(&Vt[(nt * 16 + n16) * 32], half);
        o[nt] = wmma_bf16(pfrag, vf, o[nt]);
      }
    }
  }

  // Normalize and store fp32 output
  float* ob = out + (size_t)b * kT * kHD;
#pragma unroll
  for (int r = 0; r < 8; ++r) {
    const float inv = 1.0f / lrow[r];
    const int   row = q0 + half * 8 + r;
#pragma unroll
    for (int nt = 0; nt < 8; ++nt)
      ob[(size_t)row * kHD + nt * 16 + n16] = o[nt][r] * inv;
  }
}

// ---------------------------------------------------------------------------
extern "C" void kernel_launch(void* const* d_in, const int* in_sizes, int n_in,
                              void* d_out, int out_size, void* d_ws, size_t ws_size,
                              hipStream_t stream) {
  (void)in_sizes; (void)n_in; (void)out_size; (void)ws_size;
  const float* x  = (const float*)d_in[0];
  const float* Wq = (const float*)d_in[1];
  const float* Wk = (const float*)d_in[2];
  const float* Wv = (const float*)d_in[3];
  const float* rc = (const float*)d_in[4];
  const float* rs = (const float*)d_in[5];
  // d_in[6] = additive mask: not read; causal mask is computed inline.

  unsigned short* qkv = (unsigned short*)d_ws;  // Q|K|V bf16, 3 * B*T*HD (6 MB)
  float* out = (float*)d_out;

  dim3 g1((kB * kT) / 128, 3);  // 64 x 3 blocks, 8 waves each
  qkv_rope_kernel<<<g1, 256, 0, stream>>>(x, Wq, Wk, Wv, rc, rs, qkv);

  dim3 g2(kT / 64, kB);         // 32 x 4 blocks, 4 waves each
  attention_kernel<<<g2, 128, 0, stream>>>(qkv, out);
}
